// Block_10471130268043
// MI455X (gfx1250) — compile-verified, matrix operand order checked by reading, not run on hardware
//
#include <hip/hip_runtime.h>

// ---------------------------------------------------------------------------
// Types for CDNA5 WMMA
// ---------------------------------------------------------------------------
typedef __attribute__((ext_vector_type(16))) __bf16 v16bf;
typedef __attribute__((ext_vector_type(8)))  float  v8f;

struct B32x8 { uint4 lo, hi; };

static __device__ __forceinline__ v16bf make_frag(uint4 lo, uint4 hi) {
    B32x8 t{lo, hi};
    return __builtin_bit_cast(v16bf, t);
}

static __device__ __forceinline__ v8f wmma_bf16(v16bf a, v16bf b, v8f c) {
    return __builtin_amdgcn_wmma_f32_16x16x32_bf16(
        /*neg_a=*/false, a, /*neg_b=*/false, b,
        /*c_mod=*/(short)0, c, /*reuse_a=*/false, /*reuse_b=*/false);
}

// fp32 -> bf16 (round to nearest even)
static __device__ __forceinline__ unsigned short f2bf(float f) {
    unsigned u = __builtin_bit_cast(unsigned, f);
    u += 0x7FFFu + ((u >> 16) & 1u);
    return (unsigned short)(u >> 16);
}

// CDNA5 async global->LDS copy (16 bytes/lane), tracked by ASYNCcnt.
// VDST = per-lane LDS byte address (low 32 bits of flat shared pointer),
// VADDR = per-lane 64-bit global address, GV mode (saddr = off).
static __device__ __forceinline__ void async_copy16(
    unsigned short* lds_dst, const unsigned short* gsrc)
{
    unsigned lds = (unsigned)(uintptr_t)lds_dst;
    unsigned long long ga = (unsigned long long)(uintptr_t)gsrc;
    asm volatile("global_load_async_to_lds_b128 %0, %1, off"
                 :: "v"(lds), "v"(ga) : "memory");
}

#define ASYNC_WAIT(n) asm volatile("s_wait_asynccnt " #n ::: "memory")

// ---------------------------------------------------------------------------
// Weight packing kernels (fp32 -> bf16, transposed to [N][K] for B-fragments)
// ---------------------------------------------------------------------------
__global__ __launch_bounds__(256) void pack_qkv(
    const float* __restrict__ Wq, const float* __restrict__ Wk,
    const float* __restrict__ Wv, unsigned short* __restrict__ Wt)
{
    int idx = blockIdx.x * 256 + threadIdx.x;      // over 3072*1024
    if (idx >= 3072 * 1024) return;
    int n = idx >> 10;
    int e = idx & 1023;
    const float* W = (n < 1024) ? Wq : ((n < 2048) ? Wk : Wv);
    int nn = n & 1023;
    int h = nn >> 6, d = nn & 63;
    Wt[idx] = f2bf(W[((size_t)h * 1024 + e) * 64 + d]);
}

// src [K][N] row-major fp32 -> dst [N][K] bf16
__global__ __launch_bounds__(256) void pack_t(
    const float* __restrict__ src, unsigned short* __restrict__ dst, int K, int N)
{
    int idx = blockIdx.x * 256 + threadIdx.x;
    if (idx >= K * N) return;
    int n = idx / K;
    int k = idx - n * K;
    dst[idx] = f2bf(src[(size_t)k * N + n]);
}

// ---------------------------------------------------------------------------
// LayerNorm (row of 1024) -> bf16
// ---------------------------------------------------------------------------
__global__ __launch_bounds__(256) void layernorm_bf16(
    const float* __restrict__ x, const float* __restrict__ w,
    const float* __restrict__ b, unsigned short* __restrict__ out)
{
    const int row = blockIdx.x;
    const int tid = threadIdx.x;
    const float* xr = x + (size_t)row * 1024;
    __shared__ float red[256];

    float vals[4];
    float lsum = 0.f;
#pragma unroll
    for (int i = 0; i < 4; i++) { vals[i] = xr[tid + i * 256]; lsum += vals[i]; }
    red[tid] = lsum; __syncthreads();
    for (int s = 128; s > 0; s >>= 1) {
        if (tid < s) red[tid] += red[tid + s];
        __syncthreads();
    }
    float mu = red[0] * (1.f / 1024.f);
    __syncthreads();

    float lvar = 0.f;
#pragma unroll
    for (int i = 0; i < 4; i++) { float d = vals[i] - mu; lvar += d * d; }
    red[tid] = lvar; __syncthreads();
    for (int s = 128; s > 0; s >>= 1) {
        if (tid < s) red[tid] += red[tid + s];
        __syncthreads();
    }
    float rsig = rsqrtf(red[0] * (1.f / 1024.f) + 1e-5f);

    unsigned short* orow = out + (size_t)row * 1024;
#pragma unroll
    for (int i = 0; i < 4; i++) {
        int c = tid + i * 256;
        orow[c] = f2bf((vals[i] - mu) * rsig * w[c] + b[c]);
    }
}

// ---------------------------------------------------------------------------
// Tiled bf16 WMMA GEMM with double-buffered async global->LDS staging.
// C[M][N] = A[M][K] * Bt[N][K]^T (+bias, +res, relu)
// Block tile 128x128, 8 waves (wave32), each wave: 64x32 (4x2 WMMA tiles).
// ---------------------------------------------------------------------------
template <bool OUT_BF16, bool HAS_BIAS, bool HAS_RES, bool RELU>
__global__ __launch_bounds__(256) void gemm_bf16(
    const unsigned short* __restrict__ A,   // [M][K] bf16 row-major
    const unsigned short* __restrict__ Bt,  // [N][K] bf16 row-major (pre-transposed)
    const float* __restrict__ bias,         // [N] or null
    const float* __restrict__ res,          // [M][N] fp32 or null
    void* __restrict__ Cout,                // bf16 or fp32 [M][N]
    int M, int N, int K)
{
    constexpr int BM = 128, BN = 128, BK = 32, LDT = BK + 8; // 40 halves = 80B row
    constexpr int TSZ = BM * LDT;
    __shared__ unsigned short As[2 * TSZ];
    __shared__ unsigned short Bs[2 * TSZ];

    const int tid  = threadIdx.x;
    const int lane = tid & 31;
    const int wid  = tid >> 5;
    const int wm   = wid >> 2;   // 0..1 -> 64-row half
    const int wn   = wid & 3;    // 0..3 -> 32-col quarter
    const int m0   = blockIdx.y * BM;
    const int n0   = blockIdx.x * BN;

    const int mfrag = lane & 15;
    const int kof   = (lane >> 4) * 8;

    // this thread's staging slot: row r, col chunk cc for c=0/1
    const int r0  = (tid * 2) >> 2;            // == tid/2
    const int cc0 = ((tid * 2) & 3) * 8;       // 0 or 16
    const int r1  = (tid * 2 + 1) >> 2;
    const int cc1 = ((tid * 2 + 1) & 3) * 8;   // 8 or 24

    v8f acc[4][2];
#pragma unroll
    for (int i = 0; i < 4; i++)
#pragma unroll
        for (int j = 0; j < 2; j++) acc[i][j] = {};

    const int nk = K / BK;

    // prologue: async-stage tile 0 into buffer 0 (4 async ops per wave)
    {
        unsigned short* as = As;
        unsigned short* bs = Bs;
        async_copy16(as + r0 * LDT + cc0, A  + (size_t)(m0 + r0) * K + cc0);
        async_copy16(bs + r0 * LDT + cc0, Bt + (size_t)(n0 + r0) * K + cc0);
        async_copy16(as + r1 * LDT + cc1, A  + (size_t)(m0 + r1) * K + cc1);
        async_copy16(bs + r1 * LDT + cc1, Bt + (size_t)(n0 + r1) * K + cc1);
    }

    for (int it = 0; it < nk; it++) {
        const int cur = it & 1;
        const bool has_next = (it + 1) < nk;
        if (has_next) {
            const int k0 = (it + 1) * BK;
            unsigned short* as = As + (cur ^ 1) * TSZ;
            unsigned short* bs = Bs + (cur ^ 1) * TSZ;
            async_copy16(as + r0 * LDT + cc0, A  + (size_t)(m0 + r0) * K + k0 + cc0);
            async_copy16(bs + r0 * LDT + cc0, Bt + (size_t)(n0 + r0) * K + k0 + cc0);
            async_copy16(as + r1 * LDT + cc1, A  + (size_t)(m0 + r1) * K + k0 + cc1);
            async_copy16(bs + r1 * LDT + cc1, Bt + (size_t)(n0 + r1) * K + k0 + cc1);
            ASYNC_WAIT(4);    // current tile's 4 copies done; next tile in flight
        } else {
            ASYNC_WAIT(0);
        }
        __syncthreads();      // whole tile valid across all waves

        const unsigned short* as = As + cur * TSZ;
        const unsigned short* bs = Bs + cur * TSZ;
        v16bf afr[4], bfr[2];
#pragma unroll
        for (int mi = 0; mi < 4; mi++) {
            const unsigned short* p = as + (wm * 64 + mi * 16 + mfrag) * LDT;
            afr[mi] = make_frag(*(const uint4*)(p + kof),
                                *(const uint4*)(p + 16 + kof));
        }
#pragma unroll
        for (int ni = 0; ni < 2; ni++) {
            const unsigned short* p = bs + (wn * 32 + ni * 16 + mfrag) * LDT;
            bfr[ni] = make_frag(*(const uint4*)(p + kof),
                                *(const uint4*)(p + 16 + kof));
        }
#pragma unroll
        for (int mi = 0; mi < 4; mi++)
#pragma unroll
            for (int ni = 0; ni < 2; ni++)
                acc[mi][ni] = wmma_bf16(afr[mi], bfr[ni], acc[mi][ni]);
        __syncthreads();      // all waves done reading before it gets overwritten
    }

    // epilogue: lane holds col n = lane&15, rows (lane>>4)*8 + i
#pragma unroll
    for (int mi = 0; mi < 4; mi++) {
#pragma unroll
        for (int ni = 0; ni < 2; ni++) {
            int ncol = n0 + wn * 32 + ni * 16 + (lane & 15);
            float bv = 0.f;
            if (HAS_BIAS) bv = bias[ncol];
#pragma unroll
            for (int i = 0; i < 8; i++) {
                int mrow = m0 + wm * 64 + mi * 16 + (lane >> 4) * 8 + i;
                float v = acc[mi][ni][i] + bv;
                if (HAS_RES) v += res[(size_t)mrow * N + ncol];
                if (RELU) v = v > 0.f ? v : 0.f;
                if (OUT_BF16)
                    ((unsigned short*)Cout)[(size_t)mrow * N + ncol] = f2bf(v);
                else
                    ((float*)Cout)[(size_t)mrow * N + ncol] = v;
            }
        }
    }
}

// ---------------------------------------------------------------------------
// Fused causal attention per (b, h), one wave per 16-row tile of T.
// qkv: bf16 [B*T][3072]; cols [0,1024)=Q, [1024,2048)=K, [2048,3072)=V,
//      per-head slice h*64..h*64+63.
// out: bf16 [B*T][1024] head-concatenated attention output.
// ---------------------------------------------------------------------------
__global__ __launch_bounds__(32) void attn_kernel(
    const unsigned short* __restrict__ qkv, unsigned short* __restrict__ attn)
{
    constexpr int T = 320, E3 = 3072, SLD = 336; // 336*2B row = 672B (16B mult)
    const int bx   = blockIdx.x;        // row tile 0..19
    const int h    = blockIdx.y;        // head
    const int b    = blockIdx.z;        // batch
    const int t0   = bx * 16;
    const int lane = threadIdx.x;

    __shared__ float          Sc[16 * SLD];
    __shared__ unsigned short P [16 * SLD];
    __shared__ unsigned short Vt[64 * 40];

    const size_t base = (size_t)b * T * E3;
    const int mfrag = lane & 15;
    const int kof   = (lane >> 4) * 8;

    // Q fragments for rows t0..t0+15 (two K=32 chunks over HS=64)
    v16bf qf[2];
    {
        const unsigned short* qrow = qkv + base + (size_t)(t0 + mfrag) * E3 + h * 64;
#pragma unroll
        for (int c = 0; c < 2; c++)
            qf[c] = make_frag(*(const uint4*)(qrow + c * 32 + kof),
                              *(const uint4*)(qrow + c * 32 + 16 + kof));
    }

    const float scale = 0.03125f; // E^-0.5, E=1024 (reference scales by full E)

    // scores = Q K^T for all causal s-tiles; K fragments direct from global
    for (int st = 0; st <= bx; st++) {
        v8f sc = {};
        const unsigned short* krow =
            qkv + base + (size_t)(st * 16 + mfrag) * E3 + 1024 + h * 64;
#pragma unroll
        for (int c = 0; c < 2; c++) {
            v16bf kf = make_frag(*(const uint4*)(krow + c * 32 + kof),
                                 *(const uint4*)(krow + c * 32 + 16 + kof));
            sc = wmma_bf16(qf[c], kf, sc);
        }
#pragma unroll
        for (int i = 0; i < 8; i++) {
            int m = (lane >> 4) * 8 + i;         // local t
            int s = st * 16 + (lane & 15);       // global s
            float v = sc[i] * scale;
            if (s > t0 + m) v = -3.0e38f;        // causal mask
            Sc[m * SLD + s] = v;
        }
    }
    __syncthreads();

    // softmax: lane r < 16 owns row r
    const int Send = (bx + 1) * 16;
    const int nch  = (Send + 31) / 32;           // 32-wide K chunks for P*V
    if (lane < 16) {
        const int r = lane;
        const int L = t0 + r + 1;                // valid length
        float mx = -3.0e38f;
        for (int s = 0; s < L; s++) mx = fmaxf(mx, Sc[r * SLD + s]);
        float sum = 0.f;
        for (int s = 0; s < L; s++) {
            float e = __expf(Sc[r * SLD + s] - mx);
            Sc[r * SLD + s] = e;
            sum += e;
        }
        float inv = 1.f / sum;
        for (int s = 0; s < L; s++)        P[r * SLD + s] = f2bf(Sc[r * SLD + s] * inv);
        for (int s = L; s < nch * 32; s++) P[r * SLD + s] = 0;
    }
    __syncthreads();

    // O = P V : K = s in chunks of 32, N = HS = 64 (4 WMMA n-tiles)
    v8f acc[4];
#pragma unroll
    for (int dt = 0; dt < 4; dt++) acc[dt] = {};

    for (int c = 0; c < nch; c++) {
        // stage V chunk transposed: Vt[d][s_local], lane owns s = c*32+lane
        const unsigned short* vrow =
            qkv + base + (size_t)(c * 32 + lane) * E3 + 2048 + h * 64;
#pragma unroll
        for (int d0 = 0; d0 < 64; d0 += 8) {
            uint4 v = *(const uint4*)(vrow + d0);
            const unsigned short* pv = (const unsigned short*)&v;
#pragma unroll
            for (int j = 0; j < 8; j++) Vt[(d0 + j) * 40 + lane] = pv[j];
        }
        __syncthreads();

        const unsigned short* prow = P + mfrag * SLD + c * 32;
        v16bf pf = make_frag(*(const uint4*)(prow + kof),
                             *(const uint4*)(prow + 16 + kof));
#pragma unroll
        for (int dt = 0; dt < 4; dt++) {
            const unsigned short* vb = Vt + (dt * 16 + mfrag) * 40;
            v16bf vf = make_frag(*(const uint4*)(vb + kof),
                                 *(const uint4*)(vb + 16 + kof));
            acc[dt] = wmma_bf16(pf, vf, acc[dt]);
        }
        __syncthreads();
    }

    // store: attn[b*T + t][h*64 + d]
#pragma unroll
    for (int dt = 0; dt < 4; dt++) {
        int ncol = h * 64 + dt * 16 + (lane & 15);
#pragma unroll
        for (int i = 0; i < 8; i++) {
            int t = t0 + (lane >> 4) * 8 + i;
            attn[((size_t)b * T + t) * 1024 + ncol] = f2bf(acc[dt][i]);
        }
    }
}

// ---------------------------------------------------------------------------
// Launch
// ---------------------------------------------------------------------------
extern "C" void kernel_launch(void* const* d_in, const int* in_sizes, int n_in,
                              void* d_out, int out_size, void* d_ws, size_t ws_size,
                              hipStream_t stream)
{
    (void)in_sizes; (void)n_in; (void)out_size; (void)ws_size;

    const float* x    = (const float*)d_in[0];
    const float* Wq   = (const float*)d_in[1];
    const float* Wk   = (const float*)d_in[2];
    const float* Wv   = (const float*)d_in[3];
    const float* Wo   = (const float*)d_in[4];
    const float* bo   = (const float*)d_in[5];
    const float* W1   = (const float*)d_in[6];
    const float* b1   = (const float*)d_in[7];
    const float* W2   = (const float*)d_in[8];
    const float* b2   = (const float*)d_in[9];
    const float* ln1w = (const float*)d_in[10];
    const float* ln1b = (const float*)d_in[11];
    const float* ln2w = (const float*)d_in[12];
    const float* ln2b = (const float*)d_in[13];

    const int NR = 64 * 320; // 20480 rows

    char* ws = (char*)d_ws;
    size_t off = 0;
    auto alloc = [&](size_t bytes) -> void* {
        void* p = ws + off;
        off += (bytes + 255) & ~(size_t)255;
        return p;
    };
    unsigned short* hb    = (unsigned short*)alloc((size_t)NR * 1024 * 2);
    unsigned short* h2b   = (unsigned short*)alloc((size_t)NR * 1024 * 2);
    unsigned short* qkv   = (unsigned short*)alloc((size_t)NR * 3072 * 2);
    unsigned short* attn  = (unsigned short*)alloc((size_t)NR * 1024 * 2);
    float*          x1    = (float*)         alloc((size_t)NR * 1024 * 4);
    unsigned short* ffn1  = (unsigned short*)alloc((size_t)NR * 4096 * 2);
    unsigned short* Wqkvt = (unsigned short*)alloc((size_t)3072 * 1024 * 2);
    unsigned short* Wot   = (unsigned short*)alloc((size_t)1024 * 1024 * 2);
    unsigned short* W1t   = (unsigned short*)alloc((size_t)4096 * 1024 * 2);
    unsigned short* W2t   = (unsigned short*)alloc((size_t)1024 * 4096 * 2);

    // one-time weight packs (fp32 -> bf16, [N][K])
    pack_qkv<<<(3072 * 1024) / 256, 256, 0, stream>>>(Wq, Wk, Wv, Wqkvt);
    pack_t<<<(1024 * 1024) / 256, 256, 0, stream>>>(Wo, Wot, 1024, 1024);
    pack_t<<<(1024 * 4096) / 256, 256, 0, stream>>>(W1, W1t, 1024, 4096);
    pack_t<<<(4096 * 1024) / 256, 256, 0, stream>>>(W2, W2t, 4096, 1024);

    // LN1 -> hb
    layernorm_bf16<<<NR, 256, 0, stream>>>(x, ln1w, ln1b, hb);

    // QKV: [20480,1024] x [1024,3072] -> qkv (bf16)
    gemm_bf16<true, false, false, false>
        <<<dim3(3072 / 128, NR / 128), 256, 0, stream>>>(
            hb, Wqkvt, nullptr, nullptr, qkv, NR, 3072, 1024);

    // causal attention per (tile, head, batch)
    attn_kernel<<<dim3(20, 16, 64), 32, 0, stream>>>(qkv, attn);

    // out-proj + bo + residual(x) -> x1 (fp32)
    gemm_bf16<false, true, true, false>
        <<<dim3(1024 / 128, NR / 128), 256, 0, stream>>>(
            attn, Wot, bo, x, x1, NR, 1024, 1024);

    // LN2 -> h2b
    layernorm_bf16<<<NR, 256, 0, stream>>>(x1, ln2w, ln2b, h2b);

    // FFN1 + b1 + ReLU -> ffn1 (bf16)
    gemm_bf16<true, true, false, true>
        <<<dim3(4096 / 128, NR / 128), 256, 0, stream>>>(
            h2b, W1t, b1, nullptr, ffn1, NR, 4096, 1024);

    // FFN2 + b2 + residual(x1) -> d_out (fp32)
    gemm_bf16<false, true, true, false>
        <<<dim3(1024 / 128, NR / 128), 256, 0, stream>>>(
            ffn1, W2t, b2, x1, (float*)d_out, NR, 1024, 4096);
}
